// EncoderLayer_24077586661470
// MI455X (gfx1250) — matrix operand order repaired and checked by reading, not compiled, not run on hardware
//
#include <hip/hip_runtime.h>

// ---------------------------------------------------------------------------
// MI455X (gfx1250) encoder layer: bf16 WMMA GEMMs + flash attention.
// Wave32; WMMA 16x16x32 bf16 -> f32 accum; TDM + async global->LDS staging,
// double-buffered (copy of tile k+1 overlaps compute of tile k).
// ---------------------------------------------------------------------------

#define GEMM_USE_TDM 1

typedef __attribute__((ext_vector_type(16))) __bf16    v16bf;
typedef __attribute__((ext_vector_type(8)))  float     v8f;
typedef __attribute__((ext_vector_type(8)))  unsigned  v8u;
typedef __attribute__((ext_vector_type(4)))  unsigned  v4u;

static __device__ __forceinline__ unsigned short f2bf(float f) {
  unsigned u = __float_as_uint(f);
  unsigned r = u + 0x7fffu + ((u >> 16) & 1u);   // round-to-nearest-even
  return (unsigned short)(r >> 16);
}

static __device__ __forceinline__ v8f wmma_bf16(v8u a, v8u b, v8f c) {
  return __builtin_amdgcn_wmma_f32_16x16x32_bf16(
      false, __builtin_bit_cast(v16bf, a),
      false, __builtin_bit_cast(v16bf, b),
      (short)0, c, false, false);
}

static __device__ __forceinline__ v8u mk8(v4u a, v4u b) {
  v8u r;
  r[0] = a[0]; r[1] = a[1]; r[2] = a[2]; r[3] = a[3];
  r[4] = b[0]; r[5] = b[1]; r[6] = b[2]; r[7] = b[3];
  return r;
}

// CDNA5 async DMA: 16B global->LDS, tracked by ASYNCcnt.
static __device__ __forceinline__ void async16(const unsigned short* lds_dst,
                                               const unsigned short* g_src) {
  unsigned loff = (unsigned)(unsigned long long)lds_dst;  // aperture rule
  asm volatile("global_load_async_to_lds_b128 %0, %1, off"
               :: "v"(loff), "v"(g_src) : "memory");
}
static __device__ __forceinline__ void wait_async() {
  asm volatile("s_wait_asynccnt 0" ::: "memory");
}
static __device__ __forceinline__ void lds_fence() {
  asm volatile("s_wait_dscnt 0" ::: "memory");   // wave-private LDS RAW fence
}

#if GEMM_USE_TDM
// CDNA5 Tensor Data Mover: 2D tile (tile_w x tile_h bf16) global->LDS with
// row padding (pad_interval=4 -> every 32 DWORDs=128B, pad_amount=3 -> +16B),
// giving a 144B LDS row pitch. D# per CDNA5 ISA.
static __device__ __forceinline__ void tdm_load_2d(
    const unsigned short* lds_dst, const unsigned short* g_src,
    unsigned tile_w, unsigned tile_h, unsigned row_stride_elems) {
  unsigned ldsOff = (unsigned)(unsigned long long)lds_dst;
  unsigned long long ga = (unsigned long long)g_src;
  v4u g0;
  g0[0] = 1u;                                              // count=1
  g0[1] = ldsOff;                                          // lds_addr
  g0[2] = (unsigned)(ga & 0xffffffffu);                    // global_addr lo
  g0[3] = (unsigned)((ga >> 32) & 0x01ffffffu) | (2u << 30);  // hi | type=2
  const unsigned td0 = 1u << 20, td1 = 1u << 20;           // huge dims: no OOB
  v8u g1;
  g1[0] = (1u << 16) | (1u << 20) | (4u << 22) | (3u << 25); // 2B,pad on
  g1[1] = (td0 & 0xffffu) << 16;                           // dim0 lo16
  g1[2] = (td0 >> 16) | ((td1 & 0xffffu) << 16);           // dim0 hi|dim1 lo
  g1[3] = (td1 >> 16) | (tile_w << 16);                    // dim1 hi|tile_dim0
  g1[4] = tile_h & 0xffffu;                                // tile_dim1
  g1[5] = row_stride_elems;                                // dim0_stride lo32
  g1[6] = 0u;
  g1[7] = 0u;
  asm volatile("tensor_load_to_lds %0, %1" :: "s"(g0), "s"(g1) : "memory");
}
#endif

// ---------------------------------------------------------------------------
// fp32 -> bf16 convert (row-major copy)
// ---------------------------------------------------------------------------
__global__ __launch_bounds__(256) void cvt_bf16_kernel(
    const float* __restrict__ in, unsigned short* __restrict__ out, int n) {
  int i = blockIdx.x * 256 + threadIdx.x;
  if (i < n) out[i] = f2bf(in[i]);
}

// fp32 [K][N] -> bf16 transposed [N][K] (64x64 LDS tile transpose)
__global__ __launch_bounds__(256) void cvt_t_kernel(
    const float* __restrict__ in, unsigned short* __restrict__ out,
    int K, int N) {
  __shared__ unsigned short tile[64 * 72];
  const int kb = blockIdx.y * 64, nb = blockIdx.x * 64, t = threadIdx.x;
#pragma unroll
  for (int i = 0; i < 16; ++i) {
    int u = t + i * 256, r = u >> 6, c = u & 63;
    tile[r * 72 + c] = f2bf(in[(size_t)(kb + r) * N + nb + c]);
  }
  __syncthreads();
#pragma unroll
  for (int i = 0; i < 16; ++i) {
    int u = t + i * 256, r = u >> 6, c = u & 63;
    out[(size_t)(nb + r) * K + kb + c] = tile[c * 72 + r];
  }
}

// ---------------------------------------------------------------------------
// bf16 GEMM:  C[M,N] = act(scale * (A[M,K] x BT[N,K]^T + bias) (+ res))
// Block 256 thr (8 waves), tile 128x128, BK=64; double-buffered TDM staging.
// SCL: multiply output by 0.125 (exact in bf16) -- used by the Q projection.
// WBT: store bf16 output transposed per-head ([B,H,dk,S], S=2048).
// ---------------------------------------------------------------------------
template <int RELU, int RES, int WF32, int WB16, int WBT, int SCL>
__global__ __launch_bounds__(256, 2) void gemm_bf16_kernel(
    const unsigned short* __restrict__ A,    // [M,K] bf16
    const unsigned short* __restrict__ BT,   // [N,K] bf16 (pre-transposed)
    const float* __restrict__ bias,          // [N]
    const float* __restrict__ res,           // [M,N] fp32 (if RES)
    float* __restrict__ Cf,                  // [M,N] fp32 (if WF32)
    unsigned short* __restrict__ Cb,         // bf16 out (if WB16/WBT)
    int M, int N, int K) {
  constexpr int BM = 128, BN = 128, BK = 64;
  constexpr int LDT = BK + 8;  // 72 ushorts = 144B rows (16B multiple)

  __shared__ __attribute__((aligned(16))) unsigned short As[2][BM * LDT];
  __shared__ __attribute__((aligned(16))) unsigned short Bs[2][BN * LDT];

  const int t = threadIdx.x;
  const int lane = t & 31, w = t >> 5;
  const int half = lane >> 4, ln = lane & 15;
  const int wm = w >> 1, wn = w & 1;
  const int rowBase = blockIdx.y * BM;
  const int colBase = blockIdx.x * BN;

  v8f acc[2][4];
#pragma unroll
  for (int mt = 0; mt < 2; ++mt)
#pragma unroll
    for (int nt = 0; nt < 4; ++nt)
#pragma unroll
      for (int j = 0; j < 8; ++j) acc[mt][nt][j] = 0.0f;

  auto stage = [&](int k0, int buf) {
#if GEMM_USE_TDM
    if (w == 0) {   // one wave drives the whole-tile DMA (EXEC-independent)
      tdm_load_2d(As[buf], A + (size_t)rowBase * K + k0, BK, BM, K);
      tdm_load_2d(Bs[buf], BT + (size_t)colBase * K + k0, BK, BN, K);
    }
#else
#pragma unroll
    for (int i = 0; i < 4; ++i) {
      int u = t + i * 256, r = u >> 3, c = u & 7;
      async16(&As[buf][r * LDT + c * 8],
              A + (size_t)(rowBase + r) * K + k0 + c * 8);
      async16(&Bs[buf][r * LDT + c * 8],
              BT + (size_t)(colBase + r) * K + k0 + c * 8);
    }
#endif
  };

  const int nk = K / BK;
  stage(0, 0);
  for (int kt = 0; kt < nk; ++kt) {
#if GEMM_USE_TDM
    if (w == 0) __builtin_amdgcn_s_wait_tensorcnt(0);
#else
    wait_async();
#endif
    __syncthreads();                    // tile kt visible; prior reads done
    if (kt + 1 < nk) stage((kt + 1) * BK, (kt + 1) & 1);

    const unsigned short* Ab = As[kt & 1];
    const unsigned short* Bb = Bs[kt & 1];
#pragma unroll
    for (int ks = 0; ks < BK; ks += 32) {
      v8u afr[2], bfr[4];
#pragma unroll
      for (int mt = 0; mt < 2; ++mt) {
        const v4u* rp = (const v4u*)(Ab + (wm * 32 + mt * 16 + ln) * LDT);
        int kb = (ks >> 3) + half;
        afr[mt] = mk8(rp[kb], rp[kb + 2]);
      }
#pragma unroll
      for (int nt = 0; nt < 4; ++nt) {
        const v4u* rp = (const v4u*)(Bb + (wn * 64 + nt * 16 + ln) * LDT);
        int kb = (ks >> 3) + half;
        bfr[nt] = mk8(rp[kb], rp[kb + 2]);
      }
#pragma unroll
      for (int mt = 0; mt < 2; ++mt)
#pragma unroll
        for (int nt = 0; nt < 4; ++nt)
          acc[mt][nt] = wmma_bf16(afr[mt], bfr[nt], acc[mt][nt]);
    }
    __syncthreads();                    // done reading buf before re-stage
  }

  // ---- epilogue ----
#pragma unroll
  for (int mt = 0; mt < 2; ++mt)
#pragma unroll
    for (int nt = 0; nt < 4; ++nt) {
      int col = colBase + wn * 64 + nt * 16 + ln;
      float bcol = bias[col];
#pragma unroll
      for (int r = 0; r < 8; ++r) {
        int row = rowBase + wm * 32 + mt * 16 + half * 8 + r;
        float v = acc[mt][nt][r] + bcol;
        if (SCL) v *= 0.125f;           // folded attention score scale
        if (RES) v += res[(size_t)row * N + col];
        if (RELU) v = fmaxf(v, 0.0f);
        if (WF32) Cf[(size_t)row * N + col] = v;
        if (WB16) Cb[(size_t)row * N + col] = f2bf(v);
        if (WBT)   // [B,H,dk,S]: b=row>>11, s=row&2047
          Cb[((size_t)((row >> 11) << 10) + col) * 2048 + (row & 2047)] =
              f2bf(v);
      }
    }
}

// ---------------------------------------------------------------------------
// Flash attention: block = (128 q-rows, head, batch); 8 waves x 16 rows.
// d_k=64, key tiles of 128, double-buffered async staging. Q is pre-scaled
// by 1/sqrt(d_k) in its projection; mask enters as an additive 0/-inf bias.
// ---------------------------------------------------------------------------
__global__ __launch_bounds__(256) void attention_kernel(
    const unsigned short* __restrict__ Q,   // [B,S,H*64] bf16 (pre-scaled)
    const unsigned short* __restrict__ Km,  // [B,S,H*64] bf16
    const unsigned short* __restrict__ VT,  // [B,H,64,S] bf16
    const int* __restrict__ mask,           // [B,S]
    unsigned short* __restrict__ ctx,       // [B,S,H*64] bf16 (heads merged)
    int S) {
  constexpr int DM = 1024, H = 16, QT = 128, KT = 128;
  constexpr int LDK = 72;    // K tile row stride (ushort), 144B
  constexpr int LDV = 136;   // Vt row stride (ushort), 272B
  constexpr int LDP = 136;   // per-wave P row stride

  __shared__ __attribute__((aligned(16))) unsigned short Ks[2][KT * LDK];
  __shared__ __attribute__((aligned(16))) unsigned short Vts[2][64 * LDV];
  __shared__ __attribute__((aligned(16))) unsigned short Ps[8 * 16 * LDP];
  __shared__ float mkf[2][KT];

  const int t = threadIdx.x;
  const int lane = t & 31, w = t >> 5;
  const int half = lane >> 4, ln = lane & 15;
  const int h = blockIdx.y, b = blockIdx.z;
  const int qrow0 = blockIdx.x * QT + w * 16;
  const size_t headoff = (size_t)b * S * DM + h * 64;
  const size_t vthead = ((size_t)b * H + h) * 64;

  // ---- Q A-fragments straight from global ----
  v8u qf[2];
  {
    const v4u* rp = (const v4u*)(Q + headoff + (size_t)(qrow0 + ln) * DM);
#pragma unroll
    for (int ks2 = 0; ks2 < 2; ++ks2) {
      int kb = ks2 * 4 + half;
      qf[ks2] = mk8(rp[kb], rp[kb + 2]);
    }
  }

  float m_r[8], l_r[8];
  v8f oacc[4];
#pragma unroll
  for (int r = 0; r < 8; ++r) { m_r[r] = -__builtin_inff(); l_r[r] = 0.0f; }
#pragma unroll
  for (int nt = 0; nt < 4; ++nt)
#pragma unroll
    for (int j = 0; j < 8; ++j) oacc[nt][j] = 0.0f;

  auto stage_tile = [&](int kt, int buf) {
#pragma unroll
    for (int i = 0; i < 4; ++i) {
      int u = t + i * 256;
      int rk = u >> 3, ck = u & 7;     // K tile: 128 rows x 8 chunks
      async16(&Ks[buf][rk * LDK + ck * 8],
              Km + headoff + (size_t)(kt + rk) * DM + ck * 8);
      int rv = u >> 4, cv = u & 15;    // Vt tile: 64 rows x 16 chunks
      async16(&Vts[buf][rv * LDV + cv * 8],
              VT + (vthead + rv) * S + kt + cv * 8);
    }
    if (t < KT)
      mkf[buf][t] = mask[(size_t)b * S + kt + t] ? 0.0f : -__builtin_inff();
  };

  const int ntile = S / KT;
  stage_tile(0, 0);
  for (int it = 0; it < ntile; ++it) {
    wait_async();
    __syncthreads();                    // tile `it` visible; prior reads done
    if (it + 1 < ntile) stage_tile((it + 1) * KT, (it + 1) & 1);
    const unsigned short* Ksb = Ks[it & 1];
    const unsigned short* Vtb = Vts[it & 1];
    const float* mb = mkf[it & 1];

    // ---- scores (Q pre-scaled; additive mask bias) ----
    v8f sc[8];
#pragma unroll
    for (int nt = 0; nt < 8; ++nt) {
      v8f s;
#pragma unroll
      for (int j = 0; j < 8; ++j) s[j] = 0.0f;
      const v4u* rp = (const v4u*)(Ksb + (nt * 16 + ln) * LDK);
#pragma unroll
      for (int ks2 = 0; ks2 < 2; ++ks2) {
        int kb = ks2 * 4 + half;
        v8u bf_ = mk8(rp[kb], rp[kb + 2]);
        s = wmma_bf16(qf[ks2], bf_, s);
      }
      float mbias = mb[nt * 16 + ln];
#pragma unroll
      for (int r = 0; r < 8; ++r) s[r] += mbias;
      sc[nt] = s;
    }

    // ---- online softmax (rows at (VGPR r, lane-half)) ----
    float alpha[8];
#pragma unroll
    for (int r = 0; r < 8; ++r) {
      float mx = sc[0][r];
#pragma unroll
      for (int nt = 1; nt < 8; ++nt) mx = fmaxf(mx, sc[nt][r]);
      mx = fmaxf(mx, __shfl_xor(mx, 1, 32));
      mx = fmaxf(mx, __shfl_xor(mx, 2, 32));
      mx = fmaxf(mx, __shfl_xor(mx, 4, 32));
      mx = fmaxf(mx, __shfl_xor(mx, 8, 32));
      float mn = fmaxf(m_r[r], mx);
      alpha[r] = __expf(m_r[r] - mn);
      m_r[r] = mn;
    }
#pragma unroll
    for (int r = 0; r < 8; ++r) {
      float ls = 0.0f;
#pragma unroll
      for (int nt = 0; nt < 8; ++nt) {
        float p = __expf(sc[nt][r] - m_r[r]);
        sc[nt][r] = p;
        ls += p;
      }
      ls += __shfl_xor(ls, 1, 32);
      ls += __shfl_xor(ls, 2, 32);
      ls += __shfl_xor(ls, 4, 32);
      ls += __shfl_xor(ls, 8, 32);
      l_r[r] = l_r[r] * alpha[r] + ls;
    }

    // ---- P (C-layout) -> per-wave LDS -> A-layout (wave-private) ----
    unsigned short* Pw = Ps + w * 16 * LDP;
#pragma unroll
    for (int nt = 0; nt < 8; ++nt)
#pragma unroll
      for (int r = 0; r < 8; ++r)
        Pw[(half * 8 + r) * LDP + nt * 16 + ln] = f2bf(sc[nt][r]);
    lds_fence();   // per-wave in-order LDS; fence stops compiler reordering

    // ---- O = O*alpha + P x V ----
#pragma unroll
    for (int ont = 0; ont < 4; ++ont) {
      v8f o = oacc[ont];
#pragma unroll
      for (int r = 0; r < 8; ++r) o[r] *= alpha[r];
      const v4u* vp = (const v4u*)(Vtb + (ont * 16 + ln) * LDV);
      const v4u* pp = (const v4u*)(Pw + ln * LDP);
#pragma unroll
      for (int ks4 = 0; ks4 < 4; ++ks4) {
        int ka = ks4 * 4 + half;
        v8u af = mk8(pp[ka], pp[ka + 2]);
        int kv = ks4 * 4 + half;
        v8u bf_ = mk8(vp[kv], vp[kv + 2]);
        o = wmma_bf16(af, bf_, o);
      }
      oacc[ont] = o;
    }
    __syncthreads();                    // done reading buf before re-stage
  }

  // ---- normalize and write ctx (heads merged) ----
#pragma unroll
  for (int ont = 0; ont < 4; ++ont)
#pragma unroll
    for (int r = 0; r < 8; ++r) {
      float v = oacc[ont][r] / l_r[r];
      int row = qrow0 + half * 8 + r;
      int col = ont * 16 + ln;
      ctx[headoff + (size_t)row * DM + col] = f2bf(v);
    }
}

// ---------------------------------------------------------------------------
// LayerNorm over D=1024; one block per row; writes fp32 (+optional bf16)
// ---------------------------------------------------------------------------
template <int WB16>
__global__ __launch_bounds__(256) void layernorm_kernel(
    const float* __restrict__ in, const float* __restrict__ g,
    const float* __restrict__ be, float* __restrict__ outf,
    unsigned short* __restrict__ outb, int D) {
  const int row = blockIdx.x, t = threadIdx.x;
  const float* x = in + (size_t)row * D;
  float v4[4], s = 0.0f, s2 = 0.0f;
#pragma unroll
  for (int i = 0; i < 4; ++i) {
    float v = x[t + i * 256];
    v4[i] = v; s += v; s2 += v * v;
  }
#pragma unroll
  for (int off = 16; off >= 1; off >>= 1) {
    s += __shfl_xor(s, off, 32);
    s2 += __shfl_xor(s2, off, 32);
  }
  __shared__ float rs[8], rs2[8];
  if ((t & 31) == 0) { rs[t >> 5] = s; rs2[t >> 5] = s2; }
  __syncthreads();
  s = 0.0f; s2 = 0.0f;
#pragma unroll
  for (int j = 0; j < 8; ++j) { s += rs[j]; s2 += rs2[j]; }
  float mean = s * (1.0f / 1024.0f);
  float var = s2 * (1.0f / 1024.0f) - mean * mean;
  float inv = rsqrtf(var + 1e-5f);
#pragma unroll
  for (int i = 0; i < 4; ++i) {
    int c = t + i * 256;
    float y = (v4[i] - mean) * inv * g[c] + be[c];
    outf[(size_t)row * D + c] = y;
    if (WB16) outb[(size_t)row * D + c] = f2bf(y);
  }
}

// ---------------------------------------------------------------------------
// Host orchestration
// ---------------------------------------------------------------------------
extern "C" void kernel_launch(void* const* d_in, const int* in_sizes, int n_in,
                              void* d_out, int out_size, void* d_ws,
                              size_t ws_size, hipStream_t stream) {
  const float* x        = (const float*)d_in[0];
  const int*   src_mask = (const int*)d_in[1];
  const float* w_q = (const float*)d_in[2];  const float* b_q = (const float*)d_in[3];
  const float* w_k = (const float*)d_in[4];  const float* b_k = (const float*)d_in[5];
  const float* w_v = (const float*)d_in[6];  const float* b_v = (const float*)d_in[7];
  const float* w_o = (const float*)d_in[8];  const float* b_o = (const float*)d_in[9];
  const float* w1  = (const float*)d_in[10]; const float* b1  = (const float*)d_in[11];
  const float* w2  = (const float*)d_in[12]; const float* b2  = (const float*)d_in[13];
  const float* g1  = (const float*)d_in[14]; const float* be1 = (const float*)d_in[15];
  const float* g2  = (const float*)d_in[16]; const float* be2 = (const float*)d_in[17];
  (void)in_sizes; (void)n_in; (void)out_size; (void)ws_size;

  const int B = 4, S = 2048, D = 1024, H = 16, FF = 4096;
  const int M = B * S;  // 8192

  char* ws = (char*)d_ws;
  size_t off = 0;
  auto alloc = [&](size_t bytes) -> void* {
    void* p = ws + off;
    off += (bytes + 255) & ~(size_t)255;
    return p;
  };
  unsigned short* xb   = (unsigned short*)alloc((size_t)M * D * 2);
  unsigned short* wqT  = (unsigned short*)alloc((size_t)D * D * 2);
  unsigned short* wkT  = (unsigned short*)alloc((size_t)D * D * 2);
  unsigned short* wvT  = (unsigned short*)alloc((size_t)D * D * 2);
  unsigned short* woT  = (unsigned short*)alloc((size_t)D * D * 2);
  unsigned short* w1T  = (unsigned short*)alloc((size_t)D * FF * 2);
  unsigned short* w2T  = (unsigned short*)alloc((size_t)FF * D * 2);
  unsigned short* qb   = (unsigned short*)alloc((size_t)M * D * 2);
  unsigned short* kb   = (unsigned short*)alloc((size_t)M * D * 2);
  unsigned short* vt   = (unsigned short*)alloc((size_t)M * D * 2);  // [B,H,64,S]
  unsigned short* ctxb = (unsigned short*)alloc((size_t)M * D * 2);
  float*          pre  = (float*)alloc((size_t)M * D * 4);  // pre-LN1 & pre-LN2
  float*          hf   = (float*)alloc((size_t)M * D * 4);
  unsigned short* hb   = (unsigned short*)alloc((size_t)M * D * 2);
  unsigned short* f1b  = (unsigned short*)alloc((size_t)M * FF * 2);

  cvt_bf16_kernel<<<(M * D + 255) / 256, 256, 0, stream>>>(x, xb, M * D);
  cvt_t_kernel<<<dim3(D / 64, D / 64), 256, 0, stream>>>(w_q, wqT, D, D);
  cvt_t_kernel<<<dim3(D / 64, D / 64), 256, 0, stream>>>(w_k, wkT, D, D);
  cvt_t_kernel<<<dim3(D / 64, D / 64), 256, 0, stream>>>(w_v, wvT, D, D);
  cvt_t_kernel<<<dim3(D / 64, D / 64), 256, 0, stream>>>(w_o, woT, D, D);
  cvt_t_kernel<<<dim3(FF / 64, D / 64), 256, 0, stream>>>(w1, w1T, D, FF);
  cvt_t_kernel<<<dim3(D / 64, FF / 64), 256, 0, stream>>>(w2, w2T, FF, D);

  dim3 blk(256);
  // QKV projections (Q pre-scaled by 1/8; V written pre-transposed per head)
  gemm_bf16_kernel<0, 0, 0, 1, 0, 1><<<dim3(D / 128, M / 128), blk, 0, stream>>>(
      xb, wqT, b_q, nullptr, nullptr, qb, M, D, D);
  gemm_bf16_kernel<0, 0, 0, 1, 0, 0><<<dim3(D / 128, M / 128), blk, 0, stream>>>(
      xb, wkT, b_k, nullptr, nullptr, kb, M, D, D);
  gemm_bf16_kernel<0, 0, 0, 0, 1, 0><<<dim3(D / 128, M / 128), blk, 0, stream>>>(
      xb, wvT, b_v, nullptr, nullptr, vt, M, D, D);

  // attention -> ctx bf16 (heads merged)
  attention_kernel<<<dim3(S / 128, H, B), blk, 0, stream>>>(
      qb, kb, vt, src_mask, ctxb, S);

  // output projection + residual(x) -> pre-LN1 fp32
  gemm_bf16_kernel<0, 1, 1, 0, 0, 0><<<dim3(D / 128, M / 128), blk, 0, stream>>>(
      ctxb, woT, b_o, x, pre, nullptr, M, D, D);
  // LN1 -> h fp32 + h bf16
  layernorm_kernel<1><<<M, blk, 0, stream>>>(pre, g1, be1, hf, hb, D);

  // FFN1 (relu) -> bf16
  gemm_bf16_kernel<1, 0, 0, 1, 0, 0><<<dim3(FF / 128, M / 128), blk, 0, stream>>>(
      hb, w1T, b1, nullptr, nullptr, f1b, M, FF, D);
  // FFN2 + residual(h) -> pre-LN2 fp32 (reuse `pre`)
  gemm_bf16_kernel<0, 1, 1, 0, 0, 0><<<dim3(D / 128, M / 128), blk, 0, stream>>>(
      f1b, w2T, b2, hf, pre, nullptr, M, D, FF);
  // LN2 -> output fp32
  layernorm_kernel<0><<<M, blk, 0, stream>>>(pre, g2, be2, (float*)d_out,
                                             nullptr, D);
}
